// InnerProductGraphConvolution_66846870995365
// MI455X (gfx1250) — matrix operand order repaired and checked by reading, not compiled, not run on hardware
//
#include <hip/hip_runtime.h>

typedef __attribute__((ext_vector_type(16))) __bf16 v16bf;
typedef __attribute__((ext_vector_type(8)))  __bf16 v8bf;
typedef __attribute__((ext_vector_type(8)))  float  v8f;

static __device__ __forceinline__ v8f wmma_bf16(v16bf a, v16bf b, v8f c) {
  // D = A(16x32 bf16) * B(32x16 bf16) + C(16x16 f32)
  return __builtin_amdgcn_wmma_f32_16x16x32_bf16(false, a, false, b, (short)0, c,
                                                 false, false);
}

// A fragment (16x32) from a row-major tile; `tile` points at row 0 of the 16-row tile.
// ISA layout: lanes 0-15 hold row m=lane, K in {k0..k0+7} U {k0+16..k0+23};
// lanes 16-31 hold row m=lane-16, K in {k0+8..k0+15} U {k0+24..k0+31}.
static __device__ __forceinline__ v16bf load_a_frag(const __bf16* tile, int stride,
                                                    int k0, int lane) {
  const __bf16* row = tile + (size_t)(lane & 15) * stride + k0 + (((lane >> 4) & 1) << 3);
  v8bf lo = *(const v8bf*)(row);
  v8bf hi = *(const v8bf*)(row + 16);
  return __builtin_shufflevector(lo, hi, 0, 1, 2, 3, 4, 5, 6, 7,
                                 8, 9, 10, 11, 12, 13, 14, 15);
}

// B fragment (32x16) where B(k,n) = src[n][k] (src row-major, i.e. B = src^T).
// Per lane: K in {k0+8h..k0+8h+7} U {k0+16+8h..k0+16+8h+7} (two 16-byte loads).
static __device__ __forceinline__ v16bf load_bt_frag(const __bf16* src, int stride,
                                                     int k0, int lane) {
  const __bf16* row = src + (size_t)(lane & 15) * stride + k0 + (((lane >> 4) & 1) << 3);
  v8bf lo = *(const v8bf*)(row);
  v8bf hi = *(const v8bf*)(row + 16);
  return __builtin_shufflevector(lo, hi, 0, 1, 2, 3, 4, 5, 6, 7,
                                 8, 9, 10, 11, 12, 13, 14, 15);
}

// ---------------------------------------------------------------------------
// C = bf16( A(f32, MxK row-major) @ B(f32, KxN row-major) )
// Block tile 128x128, 8 waves (256 threads); wave tile 32x64 (2x4 WMMA tiles).
// B tile staged TRANSPOSED in LDS so B fragments are contiguous b128 loads.
// ---------------------------------------------------------------------------
__global__ __launch_bounds__(256) void gemm_bf16out_kernel(
    const float* __restrict__ A, const float* __restrict__ B,
    __bf16* __restrict__ C, int M, int N, int K) {
  __shared__ __align__(16) __bf16 As[128 * 40];    // [m][k]  128 x 32 (pad 40)
  __shared__ __align__(16) __bf16 BsT[128 * 40];   // [n][k]  128 x 32 (pad 40)

  const int tid = threadIdx.x, lane = tid & 31, wave = tid >> 5;
  const int g = (lane >> 4) & 1, ln = lane & 15;
  const int m0 = blockIdx.y * 128;
  const int n0 = blockIdx.x * 128;
  const int wm = (wave & 3) * 32;
  const int wn = (wave >> 2) * 64;

  const v8f vz = {0.f, 0.f, 0.f, 0.f, 0.f, 0.f, 0.f, 0.f};
  v8f acc[2][4];
#pragma unroll
  for (int i = 0; i < 2; ++i)
#pragma unroll
    for (int j = 0; j < 4; ++j) acc[i][j] = vz;

  for (int k0 = 0; k0 < K; k0 += 32) {
    __syncthreads();
    // Stage A tile (128x32 f32 -> bf16): 1024 float4 chunks, 4 per thread.
#pragma unroll
    for (int i = 0; i < 4; ++i) {
      int c = tid + i * 256;
      int row = c >> 3, col = (c & 7) * 4;
      float4 f = *(const float4*)(A + (size_t)(m0 + row) * K + k0 + col);
      __bf16* d = &As[row * 40 + col];
      d[0] = (__bf16)f.x; d[1] = (__bf16)f.y; d[2] = (__bf16)f.z; d[3] = (__bf16)f.w;
    }
    // Stage B tile (32x128 f32) transposed -> BsT[n][k]
#pragma unroll
    for (int i = 0; i < 4; ++i) {
      int c = tid + i * 256;
      int row = c >> 5, col = (c & 31) * 4;   // row = k, col = n
      float4 f = *(const float4*)(B + (size_t)(k0 + row) * N + n0 + col);
      BsT[(col + 0) * 40 + row] = (__bf16)f.x;
      BsT[(col + 1) * 40 + row] = (__bf16)f.y;
      BsT[(col + 2) * 40 + row] = (__bf16)f.z;
      BsT[(col + 3) * 40 + row] = (__bf16)f.w;
    }
    __syncthreads();

    v16bf a0 = load_a_frag(&As[wm * 40], 40, 0, lane);
    v16bf a1 = load_a_frag(&As[(wm + 16) * 40], 40, 0, lane);
#pragma unroll
    for (int j = 0; j < 4; ++j) {
      v16bf b = load_bt_frag(&BsT[(wn + 16 * j) * 40], 40, 0, lane);
      acc[0][j] = wmma_bf16(a0, b, acc[0][j]);
      acc[1][j] = wmma_bf16(a1, b, acc[1][j]);
    }
  }

  // C-layout store: element (m,n) lives at lane (n + 16*(m>=8)), vgpr m&7.
#pragma unroll
  for (int i = 0; i < 2; ++i)
#pragma unroll
    for (int j = 0; j < 4; ++j)
#pragma unroll
      for (int r = 0; r < 8; ++r) {
        int row = m0 + wm + 16 * i + r + 8 * g;
        int col = n0 + wn + 16 * j + ln;
        C[(size_t)row * N + col] = (__bf16)acc[i][j][r];
      }
}

// ---------------------------------------------------------------------------
// dst[C][R] = transpose(src[R][C])   (bf16, LDS-tiled, 64x64 tiles)
// ---------------------------------------------------------------------------
__global__ __launch_bounds__(256) void transpose_bf16_kernel(
    const __bf16* __restrict__ src, __bf16* __restrict__ dst, int R, int Ccols) {
  __shared__ __align__(16) __bf16 Ts[64 * 72];
  const int tid = threadIdx.x;
  const int r0 = blockIdx.y * 64;
  const int c0 = blockIdx.x * 64;
#pragma unroll
  for (int i = 0; i < 2; ++i) {
    int c = tid + i * 256;                 // 512 chunks of 8 bf16
    int row = c >> 3, col = (c & 7) * 8;
    *(uint4*)&Ts[row * 72 + col] =
        *(const uint4*)(src + (size_t)(r0 + row) * Ccols + c0 + col);
  }
  __syncthreads();
#pragma unroll
  for (int i = 0; i < 2; ++i) {
    int c = tid + i * 256;
    int row = c >> 3, col = (c & 7) * 8;   // row = output row (orig col)
    __align__(16) __bf16 tmp[8];
#pragma unroll
    for (int e = 0; e < 8; ++e) tmp[e] = Ts[(col + e) * 72 + row];
    *(uint4*)(dst + (size_t)(c0 + row) * R + r0 + col) = *(const uint4*)tmp;
  }
}

// ---------------------------------------------------------------------------
// Out = softmax(Xi @ Xi^T) @ V
// Xi bf16 [8192x1024] row-major; Vt bf16 [1024x8192] = V^T (so all WMMA
// operands are contiguous per-lane loads); Out f32 [8192x1024].
// One workgroup (8 waves) owns 32 output rows; each wave owns a 128-wide
// output-column slice and a 128-wide K slice of the score GEMM. Partial
// scores are summed through LDS. Online (flash) softmax.
// ---------------------------------------------------------------------------
__global__ __launch_bounds__(256) void flash_attn_kernel(
    const __bf16* __restrict__ Xi, const __bf16* __restrict__ Vt,
    float* __restrict__ Out) {
  constexpr int NROW = 8192;
  constexpr int D = 1024;
  __shared__ __align__(16) float  Sp[8 * 1024];  // 32 KB: per-wave partial S
  __shared__ __align__(16) float  Sfin[1024];    //  4 KB: reduced S (4 tiles)
  __shared__ __align__(16) __bf16 Ps[32 * 40];   // 2.5KB: P in bf16 (pad 40)

  const int tid = threadIdx.x, lane = tid & 31, wave = tid >> 5;
  const int g = (lane >> 4) & 1, ln = lane & 15;
  const int rb = blockIdx.x * 32;  // row block
  const int wn = wave * 128;       // output-column slice
  const int kw = wave * 128;       // K slice for score GEMM

  // A fragments of the Xi row block for our K slice: loaded once, reused
  // across all 256 column-block iterations (stay in VGPRs).
  v16bf Af[2][4];
#pragma unroll
  for (int kk = 0; kk < 4; ++kk) {
    Af[0][kk] = load_a_frag(Xi + (size_t)rb * D, D, kw + kk * 32, lane);
    Af[1][kk] = load_a_frag(Xi + (size_t)(rb + 16) * D, D, kw + kk * 32, lane);
  }

  const v8f vz = {0.f, 0.f, 0.f, 0.f, 0.f, 0.f, 0.f, 0.f};
  float mrow[2][8], lrow[2][8];
  v8f O[2][8];
#pragma unroll
  for (int mt = 0; mt < 2; ++mt) {
#pragma unroll
    for (int r = 0; r < 8; ++r) { mrow[mt][r] = -__builtin_inff(); lrow[mt][r] = 0.f; }
#pragma unroll
    for (int j = 0; j < 8; ++j) O[mt][j] = vz;
  }

  for (int jb = 0; jb < NROW; jb += 32) {
    __syncthreads();

    // Partial S (32x32) over our K slice; B = Xi_colblock^T straight from L2.
    v8f s[4] = {vz, vz, vz, vz};
#pragma unroll
    for (int kk = 0; kk < 4; ++kk) {
      int k0 = kw + kk * 32;
      v16bf b0 = load_bt_frag(Xi + (size_t)jb * D, D, k0, lane);
      v16bf b1 = load_bt_frag(Xi + (size_t)(jb + 16) * D, D, k0, lane);
      s[0] = wmma_bf16(Af[0][kk], b0, s[0]);
      s[1] = wmma_bf16(Af[0][kk], b1, s[1]);
      s[2] = wmma_bf16(Af[1][kk], b0, s[2]);
      s[3] = wmma_bf16(Af[1][kk], b1, s[3]);
    }

    // V fragments for P@V: two 16B loads per lane from transposed support.
    // Issued early so the global loads overlap the LDS reduce + softmax.
    v16bf Vb[8];
#pragma unroll
    for (int j = 0; j < 8; ++j)
      Vb[j] = load_bt_frag(Vt + (size_t)(wn + 16 * j) * NROW, NROW, jb, lane);

#pragma unroll
    for (int t = 0; t < 4; ++t)
#pragma unroll
      for (int r = 0; r < 8; ++r)
        Sp[wave * 1024 + t * 256 + (r + 8 * g) * 16 + ln] = s[t][r];
    __syncthreads();

    // Cross-wave reduce: each thread sums one float4 across the 8 partials.
    {
      float4 a = make_float4(0.f, 0.f, 0.f, 0.f);
#pragma unroll
      for (int w = 0; w < 8; ++w) {
        float4 p = *(const float4*)&Sp[w * 1024 + tid * 4];
        a.x += p.x; a.y += p.y; a.z += p.z; a.w += p.w;
      }
      *(float4*)&Sfin[tid * 4] = a;
    }
    __syncthreads();

    // Online softmax (every wave redundantly, identical results -> identical
    // per-row alpha lands in the lanes that hold that row's O accumulators).
    float p[2][2][8], alpha[2][8];
#pragma unroll
    for (int mt = 0; mt < 2; ++mt) {
#pragma unroll
      for (int r = 0; r < 8; ++r) {
        float s0 = Sfin[(mt * 2 + 0) * 256 + (r + 8 * g) * 16 + ln];
        float s1 = Sfin[(mt * 2 + 1) * 256 + (r + 8 * g) * 16 + ln];
        float mx = fmaxf(s0, s1);
#pragma unroll
        for (int off = 8; off >= 1; off >>= 1)
          mx = fmaxf(mx, __shfl_xor(mx, off, 16));
        float mn = fmaxf(mrow[mt][r], mx);
        float al = __expf(mrow[mt][r] - mn);
        float p0 = __expf(s0 - mn);
        float p1 = __expf(s1 - mn);
        float rs = p0 + p1;
#pragma unroll
        for (int off = 8; off >= 1; off >>= 1)
          rs += __shfl_xor(rs, off, 16);
        lrow[mt][r] = lrow[mt][r] * al + rs;
        mrow[mt][r] = mn;
        alpha[mt][r] = al;
        p[mt][0][r] = p0;
        p[mt][1][r] = p1;
      }
    }
    // One wave writes P (C-layout -> LDS) for the A-layout re-gather.
    if (wave == 0) {
#pragma unroll
      for (int mt = 0; mt < 2; ++mt)
#pragma unroll
        for (int nt = 0; nt < 2; ++nt)
#pragma unroll
          for (int r = 0; r < 8; ++r)
            Ps[(mt * 16 + r + 8 * g) * 40 + nt * 16 + ln] = (__bf16)p[mt][nt][r];
    }
    // Rescale output accumulators by alpha.
#pragma unroll
    for (int mt = 0; mt < 2; ++mt)
#pragma unroll
      for (int j = 0; j < 8; ++j)
#pragma unroll
        for (int r = 0; r < 8; ++r) O[mt][j][r] *= alpha[mt][r];

    // Prefetch next column block while we finish this one.
    if (jb + 32 < NROW) {
      __builtin_prefetch(Xi + (size_t)(jb + 32 + ln) * D + kw, 0, 0);
      __builtin_prefetch(Vt + (size_t)(wn + ln) * NROW + jb + 32, 0, 0);
    }
    __syncthreads();

    // O += P(32x32) @ V(32 x our 128-col slice)
    v16bf pa0 = load_a_frag(Ps, 40, 0, lane);
    v16bf pa1 = load_a_frag(Ps + 16 * 40, 40, 0, lane);
#pragma unroll
    for (int j = 0; j < 8; ++j) {
      O[0][j] = wmma_bf16(pa0, Vb[j], O[0][j]);
      O[1][j] = wmma_bf16(pa1, Vb[j], O[1][j]);
    }
  }

  // Epilogue: divide by the softmax denominator and store fp32.
#pragma unroll
  for (int mt = 0; mt < 2; ++mt)
#pragma unroll
    for (int j = 0; j < 8; ++j)
#pragma unroll
      for (int r = 0; r < 8; ++r) {
        int row = rb + mt * 16 + r + 8 * g;
        int col = wn + 16 * j + ln;
        Out[(size_t)row * D + col] = O[mt][j][r] / lrow[mt][r];
      }
}

extern "C" void kernel_launch(void* const* d_in, const int* in_sizes, int n_in,
                              void* d_out, int out_size, void* d_ws, size_t ws_size,
                              hipStream_t stream) {
  (void)in_sizes; (void)n_in; (void)out_size; (void)ws_size;
  const float* x         = (const float*)d_in[0];
  const float* weight    = (const float*)d_in[1];
  const float* transferi = (const float*)d_in[2];
  // d_in[3] (transferj) is unused by the reference.
  float* out = (float*)d_out;

  __bf16* xi   = (__bf16*)d_ws;                      // 8192x1024 bf16 (16 MB)
  __bf16* sup  = xi + (size_t)8192 * 1024;           // 8192x1024 bf16 (16 MB)
  __bf16* supT = sup + (size_t)8192 * 1024;          // 1024x8192 bf16 (16 MB)

  dim3 blk(256);
  // xi = bf16(x @ transferi); support = bf16(x @ weight)
  gemm_bf16out_kernel<<<dim3(8, 64), blk, 0, stream>>>(x, transferi, xi, 8192, 1024, 1024);
  gemm_bf16out_kernel<<<dim3(8, 64), blk, 0, stream>>>(x, weight, sup, 8192, 1024, 1024);
  // supT = support^T (so P@V operands are contiguous per-lane loads)
  transpose_bf16_kernel<<<dim3(16, 128), blk, 0, stream>>>(sup, supT, 8192, 1024);
  // out = softmax(xi @ xi^T) @ support
  flash_attn_kernel<<<dim3(256), blk, 0, stream>>>(xi, supT, out);
}
